// ConcatPoolingFusion_82446192214445
// MI455X (gfx1250) — matrix operand order repaired and checked by reading, not compiled, not run on hardware
//
#include <hip/hip_runtime.h>

// ---------------------------------------------------------------------------
// ConcatPoolingFusion on MI455X (gfx1250):
//   out[b,l,e] = x1[b,l,:]·W1[e,:]  +  (mean_l x2[b,l,:])·W2[e,:]  +  b[e]
//
// Pipeline:
//   1) pre-convert x1 -> bf16 (Xb) and W1 -> bf16 (Wb) once (workspace)
//   2) column-sum x2 (two-stage, deterministic) -> xbar
//   3) bias2[b,e] = xbar·W2ᵀ/2048 + b
//   4) main GEMM: bf16 WMMA, double-buffered LDS fed by
//      GLOBAL_LOAD_ASYNC_TO_LDS_B128 (ASYNCcnt pipeline) when available.
// ---------------------------------------------------------------------------

typedef __attribute__((ext_vector_type(16))) __bf16 v16bf;
typedef __attribute__((ext_vector_type(8)))  __bf16 v8bf;
typedef __attribute__((ext_vector_type(8)))  float  v8f;
typedef __attribute__((ext_vector_type(4)))  int    v4i;

#define D1 1024
#define D2 1024
#define D3 1024
#define NB 8
#define L  2048
#define M_TOTAL (NB * L)   // 16384

#define AS1 __attribute__((address_space(1)))
#define AS3 __attribute__((address_space(3)))

#if __has_builtin(__builtin_amdgcn_global_load_async_to_lds_b128)
#define HAVE_ASYNC_LDS 1
#else
#define HAVE_ASYNC_LDS 0
#endif

#if HAVE_ASYNC_LDS
#if __has_builtin(__builtin_amdgcn_s_wait_asynccnt)
#define WAIT_ASYNC(n) __builtin_amdgcn_s_wait_asynccnt(n)
#else
#define WAIT_ASYNC(n) asm volatile("s_wait_asynccnt %0" ::"n"(n) : "memory")
#endif
#else
#define WAIT_ASYNC(n) ((void)0)
#endif

// round-to-nearest-even f32 -> bf16 (bit pattern)
__device__ __forceinline__ unsigned f2bf(float f) {
  union { float f; unsigned u; } v; v.f = f;
  unsigned u = v.u;
  u += 0x7FFFu + ((u >> 16) & 1u);
  return u >> 16;
}

// 16-byte global -> LDS copy: async (no VGPR transit) when available
__device__ __forceinline__ void copy16_g2l(const unsigned short* g, unsigned short* l) {
#if HAVE_ASYNC_LDS
  __builtin_amdgcn_global_load_async_to_lds_b128(
      (AS1 v4i*)g, (AS3 v4i*)l, 0, 0);
#else
  *(uint4*)l = *(const uint4*)g;
#endif
}

// ---------------------------------------------------------------------------
// Pre-conversion kernels: f32 -> bf16 once, so the GEMM streams raw bf16.
// ---------------------------------------------------------------------------
__global__ __launch_bounds__(256) void conv_x1_bf16(
    const float* __restrict__ x, unsigned short* __restrict__ xb) {
  const size_t g = ((size_t)blockIdx.x * 256 + threadIdx.x) * 8;
  float4 a = *(const float4*)(x + g);
  float4 b = *(const float4*)(x + g + 4);
  uint4 o;
  o.x = f2bf(a.x) | (f2bf(a.y) << 16);
  o.y = f2bf(a.z) | (f2bf(a.w) << 16);
  o.z = f2bf(b.x) | (f2bf(b.y) << 16);
  o.w = f2bf(b.z) | (f2bf(b.w) << 16);
  *(uint4*)(xb + g) = o;
}

__global__ __launch_bounds__(256) void conv_w1_bf16(
    const float* __restrict__ W, unsigned short* __restrict__ wb) {
  const size_t g = ((size_t)blockIdx.x * 256 + threadIdx.x) * 8;  // elem in 1024x1024
  const size_t row = g >> 10, col = g & 1023;
  const float* p = W + row * (D1 + D2) + col;    // W1 = cols 0..1023
  float4 a = *(const float4*)(p);
  float4 b = *(const float4*)(p + 4);
  uint4 o;
  o.x = f2bf(a.x) | (f2bf(a.y) << 16);
  o.y = f2bf(a.z) | (f2bf(a.w) << 16);
  o.z = f2bf(b.x) | (f2bf(b.y) << 16);
  o.w = f2bf(b.z) | (f2bf(b.w) << 16);
  *(uint4*)(wb + g) = o;
}

// ---------------------------------------------------------------------------
// x2 column sums (deterministic two-stage reduction)
// ---------------------------------------------------------------------------
__global__ __launch_bounds__(256) void x2_partial_sum(
    const float* __restrict__ x2, float* __restrict__ partial) {
  const int d     = blockIdx.x * 256 + threadIdx.x;
  const int b     = blockIdx.y;
  const int chunk = blockIdx.z;
  const float* p = x2 + ((size_t)b * L + (size_t)chunk * 128) * D2 + d;
  float s = 0.f;
#pragma unroll 8
  for (int i = 0; i < 128; ++i) s += p[(size_t)i * D2];
  partial[((size_t)b * 16 + chunk) * D2 + d] = s;
}

__global__ __launch_bounds__(256) void x2_final_sum(
    const float* __restrict__ partial, float* __restrict__ xbar) {
  const int d = blockIdx.x * 256 + threadIdx.x;
  const int b = blockIdx.y;
  float s = 0.f;
#pragma unroll
  for (int c = 0; c < 16; ++c) s += partial[((size_t)b * 16 + c) * D2 + d];
  xbar[(size_t)b * D2 + d] = s;
}

// ---------------------------------------------------------------------------
// bias2[b,e] = (xbar[b,:]·W2[e,:]) / L + bias[e]   (wave32 shuffle reduction)
// ---------------------------------------------------------------------------
__global__ __launch_bounds__(256) void fused_bias(
    const float* __restrict__ xbar, const float* __restrict__ W,
    const float* __restrict__ bvec, float* __restrict__ bias2) {
  const int e    = blockIdx.x;
  const int bsub = threadIdx.x >> 5;
  const int lane = threadIdx.x & 31;
  const float* wrow = W + (size_t)e * (D1 + D2) + D1;
  const float* xr   = xbar + (size_t)bsub * D2;
  float s = 0.f;
#pragma unroll 4
  for (int d = lane; d < D2; d += 32) s += xr[d] * wrow[d];
#pragma unroll
  for (int off = 16; off > 0; off >>= 1) s += __shfl_xor(s, off, 32);
  if (lane == 0)
    bias2[(size_t)bsub * D3 + e] = s * (1.0f / (float)L) + bvec[e];
}

// ---------------------------------------------------------------------------
// Main GEMM: block tile 128x128, stage BK=64, double-buffered LDS,
// async global->LDS copies, 16 v_wmma_f32_16x16x32_bf16 per stage per wave.
// ---------------------------------------------------------------------------
#define BM 128
#define BN 128
#define BKE 64          // K elements per pipeline stage
#define STRIDE 72       // ushorts per LDS row (128B data + 16B pad = 144B)

union Frag16 { v16bf v; v8bf h[2]; };

__global__ __launch_bounds__(256) void gemm_bf16_wmma(
    const unsigned short* __restrict__ Xb,   // (16384, 1024) bf16
    const unsigned short* __restrict__ Wb,   // (1024, 1024) bf16
    const float* __restrict__ bias2,         // (8, 1024)
    float* __restrict__ out) {               // (16384, 1024) f32
  __shared__ unsigned short lA[2][BM * STRIDE];   // 2 x 18 KB
  __shared__ unsigned short lB[2][BN * STRIDE];   // 2 x 18 KB

  const int tid   = threadIdx.x;
  const int tileN = blockIdx.x * BN;
  const int tileM = blockIdx.y * BM;
  const int batch = tileM >> 11;        // 128 | 2048 -> one batch per tile

  const int wave  = tid >> 5;
  const int lane  = tid & 31;
  const int waveM = wave & 3;           // 4 waves along M (32 rows each)
  const int waveN = wave >> 2;          // 2 waves along N (64 cols each)
  const int rr    = lane & 15;
  const int hf    = lane >> 4;

  v8f acc[2][4] = {};

  // issue one pipeline stage: 4+4 16B async chunks per thread (8/wave)
  auto issue = [&](int buf, int k0) {
#pragma unroll
    for (int i = 0; i < 4; ++i) {
      const int c   = tid + 256 * i;    // 0..1023 16B chunks
      const int row = c >> 3;           // 0..127
      const int c8  = c & 7;            // 16B chunk within 128B row
      copy16_g2l(Xb + (size_t)(tileM + row) * D1 + k0 + c8 * 8,
                 &lA[buf][row * STRIDE + c8 * 8]);
      copy16_g2l(Wb + (size_t)(tileN + row) * D1 + k0 + c8 * 8,
                 &lB[buf][row * STRIDE + c8 * 8]);
    }
  };

  issue(0, 0);
  for (int k0 = 0; k0 < D1; k0 += BKE) {
    const int buf = (k0 >> 6) & 1;
    const bool has_next = (k0 + BKE) < D1;
    if (has_next) {
      issue(buf ^ 1, k0 + BKE);
      WAIT_ASYNC(8);    // the 8 just-issued may remain; current stage is done
    } else {
      WAIT_ASYNC(0);
    }
    __syncthreads();

#pragma unroll
    for (int kc = 0; kc < 2; ++kc) {    // two K=32 WMMA chunks per stage
      Frag16 af[2];
#pragma unroll
      for (int mi = 0; mi < 2; ++mi) {
        const unsigned short* base =
            &lA[buf][(waveM * 32 + mi * 16 + rr) * STRIDE + kc * 32 + hf * 8];
        af[mi].h[0] = *(const v8bf*)base;          // K = kc*32 + hf*8 + 0..7
        af[mi].h[1] = *(const v8bf*)(base + 16);   // K = ... + 16..23
      }
      Frag16 bfr[4];
#pragma unroll
      for (int ni = 0; ni < 4; ++ni) {
        const unsigned short* base =
            &lB[buf][(waveN * 64 + ni * 16 + rr) * STRIDE + kc * 32 + hf * 8];
        bfr[ni].h[0] = *(const v8bf*)base;         // K = kc*32 + hf*8 + 0..7
        bfr[ni].h[1] = *(const v8bf*)(base + 16);  // K = ... + 16..23
      }
#pragma unroll
      for (int mi = 0; mi < 2; ++mi)
#pragma unroll
        for (int ni = 0; ni < 4; ++ni)
          acc[mi][ni] = __builtin_amdgcn_wmma_f32_16x16x32_bf16(
              false, af[mi].v, false, bfr[ni].v,
              (short)0, acc[mi][ni], false, false);
    }
    __syncthreads();
  }

  // epilogue: fused bias add, f32 store
#pragma unroll
  for (int ni = 0; ni < 4; ++ni) {
    const int colg = tileN + waveN * 64 + ni * 16 + rr;
    const float bv = bias2[(size_t)batch * D3 + colg];
#pragma unroll
    for (int mi = 0; mi < 2; ++mi) {
      const int rowg0 = tileM + waveM * 32 + mi * 16 + hf * 8;
#pragma unroll
      for (int q = 0; q < 8; ++q)
        out[(size_t)(rowg0 + q) * D3 + colg] = acc[mi][ni][q] + bv;
    }
  }
}

// ---------------------------------------------------------------------------
extern "C" void kernel_launch(void* const* d_in, const int* in_sizes, int n_in,
                              void* d_out, int out_size, void* d_ws, size_t ws_size,
                              hipStream_t stream) {
  const float* x1   = (const float*)d_in[0];  // (8, 2048, 1024)
  const float* x2   = (const float*)d_in[1];  // (8, 2048, 1024)
  const float* W    = (const float*)d_in[2];  // (1024, 2048)
  const float* bvec = (const float*)d_in[3];  // (1024,)
  float* out = (float*)d_out;                 // (8, 2048, 1024)

  float* ws      = (float*)d_ws;
  float* partial = ws;                        // 131072 f32
  float* xbar    = ws + 131072;               // 8192 f32
  float* bias2   = ws + 131072 + 8192;        // 8192 f32
  unsigned short* Xb = (unsigned short*)(ws + 147456);        // 16M bf16 (32MB)
  unsigned short* Wb = Xb + (size_t)M_TOTAL * D1;             // 1M bf16 (2MB)

  conv_x1_bf16  <<<dim3((M_TOTAL * D1) / 2048), 256, 0, stream>>>(x1, Xb);
  conv_w1_bf16  <<<dim3((D3 * D1) / 2048),      256, 0, stream>>>(W, Wb);
  x2_partial_sum<<<dim3(D2 / 256, NB, 16),      256, 0, stream>>>(x2, partial);
  x2_final_sum  <<<dim3(D2 / 256, NB),          256, 0, stream>>>(partial, xbar);
  fused_bias    <<<dim3(D3),                    256, 0, stream>>>(xbar, W, bvec, bias2);
  gemm_bf16_wmma<<<dim3(D3 / BN, M_TOTAL / BM), 256, 0, stream>>>(Xb, Wb, bias2, out);
}
